// hypercorre_topk2_31636729102628
// MI455X (gfx1250) — matrix operand order repaired and checked by reading, not compiled, not run on hardware
//
#include <hip/hip_runtime.h>
#include <hip/hip_bf16.h>
#include <math.h>

// ---------------------------------------------------------------------------
// Types for CDNA5 WMMA
// ---------------------------------------------------------------------------
typedef __attribute__((ext_vector_type(16))) __bf16 v16bf;
typedef __attribute__((ext_vector_type(8)))  float  v8f;

#define WW 64  // fixed innermost width of every tensor in this model

// A-gather LDS offset for im2col K index kk (compile-time folded).
// Returns offset to be added to (w0+n); invalid K -> zero-pad region base.
template <int CIN, int KSZ, int LW, int KTAP>
__host__ __device__ constexpr int a_off(int kk)
{
  if (kk >= KTAP) return CIN * KSZ * LW;  // zero region (64 zeroed floats)
  int ci  = kk / (KSZ * KSZ);
  int rem = kk - ci * (KSZ * KSZ);
  int dy  = rem / KSZ;
  int dx  = rem - dy * KSZ;
  return (ci * KSZ + dy) * LW + dx;
}

// ---------------------------------------------------------------------------
// Conv2d (stride 1, pad k/2) + bias as im2col GEMM on the matrix pipe.
// One wave per (b,h) output row:
//   - stage zero-padded Cin x k x (64+2*pad) float window + weights into LDS
//   - build B (weight) fragments once (branchless LDS gathers)
//   - 4 M-tiles of 16 w-positions: branchless A gather, nK x wmma, 2x b128 st
// Emits v_wmma_f32_16x16x32_bf16 (fp32 accumulate), fully unrolled.
// ---------------------------------------------------------------------------
template <int CIN, int COUT, int KSZ>
__global__ __launch_bounds__(32) void conv_row_wmma(
    const float* __restrict__ src, float* __restrict__ dst,
    const float* __restrict__ wgt, const float* __restrict__ bias, int H)
{
  constexpr int PAD   = KSZ / 2;
  constexpr int LW    = WW + 2 * PAD;           // padded row width in LDS
  constexpr int KTAP  = CIN * KSZ * KSZ;        // im2col reduction length
  constexpr int NK    = (KTAP + 31) / 32;       // K-chunks of 32
  constexpr int ZBASE = CIN * KSZ * LW;         // zero region: 64 floats
  constexpr int WBASE = ZBASE + 64;             // staged weights
  constexpr int SMEMN = WBASE + COUT * KTAP;

  __shared__ float smem[SMEMN];

  const int lane = threadIdx.x;                 // 0..31, one wave per block
  const int lh   = lane >> 4;                   // lane half
  const int n    = lane & 15;                   // A: pixel M ; B/D: column N

  const int row = blockIdx.x;                   // Bn * H rows
  const int h   = row % H;
  const int b   = row / H;
  const float* sbase = src + (size_t)b * CIN * H * WW;

  // prefetch next row (global_prefetch_b8)
  {
    int hn = (h + 1 < H) ? (h + 1) : h;
    __builtin_prefetch(sbase + (size_t)hn * WW, 0, 1);
  }

  // ---- stage zero-padded input window (no div/mod; row check is uniform)
#pragma unroll
  for (int ci = 0; ci < CIN; ++ci) {
#pragma unroll
    for (int dy = 0; dy < KSZ; ++dy) {
      int yy = h + dy - PAD;
      bool rowok = (yy >= 0) && (yy < H);
      const float* rp = sbase + ((size_t)ci * H + yy) * WW;
      for (int x = lane; x < LW; x += 32) {
        int xx = x - PAD;
        float v = 0.0f;
        if (rowok && xx >= 0 && xx < WW) v = rp[xx];
        smem[(ci * KSZ + dy) * LW + x] = v;
      }
    }
  }
  // zero region (A/B out-of-range slots land here)
  smem[ZBASE + lane] = 0.0f;
  smem[ZBASE + 32 + lane] = 0.0f;
  // stage weights
  for (int i = lane; i < COUT * KTAP; i += 32) smem[WBASE + i] = wgt[i];
  __syncthreads();

  // ---- B fragments (weights), built once per row, branchless.
  // B 32x16 in this code's contiguous k order (A matches): lanes 0-15 hold K=0..15, lanes 16-31 K=16..31; column N=n;
  // VGPR v, half hh -> K = 16*lh + 2v + hh
  const int  nKT  = WBASE + n * KTAP;
  const bool ncol = (n < COUT);
  v16bf bfrag[NK];
#pragma unroll
  for (int kc = 0; kc < NK; ++kc) {
#pragma unroll
    for (int v = 0; v < 8; ++v) {
#pragma unroll
      for (int hh = 0; hh < 2; ++hh) {
        const int kb0 = kc * 32 + v * 2 + hh;        // lh = 0
        const int kb1 = kb0 + 16;                    // lh = 1
        int off0 = (kb0 < KTAP) ? nKT + kb0 : ZBASE;
        int off1 = (kb1 < KTAP) ? nKT + kb1 : ZBASE;
        int off  = lh ? off1 : off0;
        off      = ncol ? off : ZBASE;
        bfrag[kc][v * 2 + hh] = (__bf16)smem[off];
      }
    }
  }

  const float bn = ncol ? bias[n] : 0.0f;

  // ---- 4 M-tiles of 16 consecutive w positions
#pragma unroll
  for (int tile = 0; tile < 4; ++tile) {
    const int w0n = tile * 16 + n;
    v8f acc;
#pragma unroll
    for (int r = 0; r < 8; ++r) acc[r] = bn;

#pragma unroll
    for (int kc = 0; kc < NK; ++kc) {
      // A 16x32 in this code's contiguous k order (same as B): lane holds M=n; VGPR v, half hh ->
      // K = 16*lh + 2v + hh
      v16bf afrag;
#pragma unroll
      for (int v = 0; v < 8; ++v) {
#pragma unroll
        for (int hh = 0; hh < 2; ++hh) {
          const int kl0 = v * 2 + hh;
          constexpr int KK = KSZ * KSZ;  (void)KK;
          const int c0 = a_off<CIN, KSZ, LW, KTAP>(kc * 32 + kl0);      // const
          const int c1 = a_off<CIN, KSZ, LW, KTAP>(kc * 32 + kl0 + 16); // const
          const int off = (lh ? c1 : c0) + w0n;  // cndmask(const,const)+add
          afrag[v * 2 + hh] = (__bf16)smem[off]; // ds_load_b32 + cvt
        }
      }
      acc = __builtin_amdgcn_wmma_f32_16x16x32_bf16(
          false, afrag, false, bfrag[kc], (short)0, acc, false, false);
    }

    // D layout: lane L: N=L%16, VGPR r: M=r+8*(L/16) -> w=w0+M (8 contiguous)
    if (ncol) {
      float* dbase = dst + (((size_t)b * COUT + n) * H + h) * WW +
                     (w0n - n) + 8 * lh;
      float4 lo4 = make_float4(acc[0], acc[1], acc[2], acc[3]);
      float4 hi4 = make_float4(acc[4], acc[5], acc[6], acc[7]);
      *(float4*)(dbase + 0) = lo4;   // global_store_b128 x2 (clause)
      *(float4*)(dbase + 4) = hi4;
    }
  }
}

// ---------------------------------------------------------------------------
// GroupNorm(1 group) stats: deterministic two-pass reduction
// ---------------------------------------------------------------------------
#define NCHUNK 64

__global__ void gn_reduce1(const float* __restrict__ x, float* __restrict__ part,
                           int per_sample)
{
  int b = blockIdx.x >> 6;           // / NCHUNK
  int c = blockIdx.x & (NCHUNK - 1);
  size_t base = (size_t)b * per_sample;
  int len = (per_sample + NCHUNK - 1) / NCHUNK;
  int s = c * len;
  int e = s + len; if (e > per_sample) e = per_sample;
  float s1 = 0.f, s2 = 0.f;
  for (int i = s + (int)threadIdx.x; i < e; i += 256) {
    float v = x[base + i];
    s1 += v; s2 += v * v;
  }
  __shared__ float a1[256], a2[256];
  a1[threadIdx.x] = s1; a2[threadIdx.x] = s2;
  __syncthreads();
  for (int st = 128; st > 0; st >>= 1) {
    if ((int)threadIdx.x < st) {
      a1[threadIdx.x] += a1[threadIdx.x + st];
      a2[threadIdx.x] += a2[threadIdx.x + st];
    }
    __syncthreads();
  }
  if (threadIdx.x == 0) {
    part[(size_t)(b * NCHUNK + c) * 2 + 0] = a1[0];
    part[(size_t)(b * NCHUNK + c) * 2 + 1] = a2[0];
  }
}

__global__ void gn_reduce2(const float* __restrict__ part, float* __restrict__ stats,
                           int per_sample)
{
  int b = blockIdx.x;
  int t = threadIdx.x;  // 64 threads
  __shared__ float a1[64], a2[64];
  a1[t] = part[(size_t)(b * NCHUNK + t) * 2 + 0];
  a2[t] = part[(size_t)(b * NCHUNK + t) * 2 + 1];
  __syncthreads();
  for (int st = 32; st > 0; st >>= 1) {
    if (t < st) { a1[t] += a1[t + st]; a2[t] += a2[t + st]; }
    __syncthreads();
  }
  if (t == 0) {
    float inv = 1.0f / (float)per_sample;
    float mu  = a1[0] * inv;
    float var = a2[0] * inv - mu * mu;
    if (var < 0.f) var = 0.f;
    stats[b * 2 + 0] = mu;
    stats[b * 2 + 1] = rsqrtf(var + 1e-5f);
  }
}

// per_sample and H*W are powers of two for every layer -> shift decode
__global__ void gn_apply(float* __restrict__ x, const float* __restrict__ stats,
                         const float* __restrict__ g, const float* __restrict__ beta,
                         int log2HW, int log2per, int Cmask, size_t total)
{
  size_t i = (size_t)blockIdx.x * 256 + threadIdx.x;
  if (i >= total) return;
  size_t b = i >> log2per;
  int c = (int)((i >> log2HW) & (size_t)Cmask);
  float mu = stats[b * 2 + 0], rs = stats[b * 2 + 1];
  float v = (x[i] - mu) * rs * g[c] + beta[c];
  x[i] = v > 0.f ? v : 0.f;
}

// ---------------------------------------------------------------------------
// Bilinear align_corners interp along H (W identical -> identity) + add.
// C == 2 and Ho a power of two for both call sites -> shift decode.
// ---------------------------------------------------------------------------
__global__ void interp_add_kernel(const float* __restrict__ lo,
                                  const float* __restrict__ addv,
                                  float* __restrict__ out,
                                  int Hi, int Ho, int log2Ho, size_t total)
{
  size_t i = (size_t)blockIdx.x * 256 + threadIdx.x;
  if (i >= total) return;
  int w  = (int)(i & (WW - 1));
  int ho = (int)((i >> 6) & (size_t)(Ho - 1));
  int c  = (int)((i >> (6 + log2Ho)) & 1);
  int b  = (int)(i >> (7 + log2Ho));
  float scale = (float)(Hi - 1) / (float)(Ho - 1);
  float cc = (float)ho * scale;
  int y0 = (int)floorf(cc);
  if (y0 < 0) y0 = 0; if (y0 > Hi - 1) y0 = Hi - 1;
  int y1 = y0 + 1; if (y1 > Hi - 1) y1 = Hi - 1;
  float wy = cc - (float)y0;
  size_t base = ((size_t)b * 2 + c) * Hi;
  float v = lo[(base + y0) * WW + w] * (1.f - wy) + lo[(base + y1) * WW + w] * wy;
  out[i] = v + addv[i];
}

// ---------------------------------------------------------------------------
// Host orchestration
// ---------------------------------------------------------------------------
struct LayerP { const float *w, *b, *g, *beta; };

static int ilog2(int v) { int s = 0; while ((1 << s) < v) ++s; return s; }

static void run_layer(const float* src, float* dst, const LayerP& lp,
                      int Bn, int Cin, int Cout, int H, int ksz,
                      float* partials, float* stats, hipStream_t stream)
{
  int blocks = Bn * H;  // one wave per output row
  if      (Cin == 3 && Cout == 1 && ksz == 3)
    conv_row_wmma<3,1,3><<<blocks, 32, 0, stream>>>(src, dst, lp.w, lp.b, H);
  else if (Cin == 1 && Cout == 2 && ksz == 3)
    conv_row_wmma<1,2,3><<<blocks, 32, 0, stream>>>(src, dst, lp.w, lp.b, H);
  else if (Cin == 3 && Cout == 1 && ksz == 5)
    conv_row_wmma<3,1,5><<<blocks, 32, 0, stream>>>(src, dst, lp.w, lp.b, H);
  else if (Cin == 2 && Cout == 2 && ksz == 3)
    conv_row_wmma<2,2,3><<<blocks, 32, 0, stream>>>(src, dst, lp.w, lp.b, H);
  else if (Cin == 2 && Cout == 1 && ksz == 3)
    conv_row_wmma<2,1,3><<<blocks, 32, 0, stream>>>(src, dst, lp.w, lp.b, H);

  int per = Cout * H * WW;
  gn_reduce1<<<Bn * NCHUNK, 256, 0, stream>>>(dst, partials, per);
  gn_reduce2<<<Bn, 64, 0, stream>>>(partials, stats, per);
  size_t total = (size_t)Bn * per;
  gn_apply<<<(unsigned)((total + 255) / 256), 256, 0, stream>>>(
      dst, stats, lp.g, lp.beta, ilog2(H * WW), ilog2(per), Cout - 1, total);
}

extern "C" void kernel_launch(void* const* d_in, const int* in_sizes, int n_in,
                              void* d_out, int out_size, void* d_ws, size_t ws_size,
                              hipStream_t stream)
{
  const int Bn = 16;
  // ---- identify activation inputs by unique element counts
  const float *hc4 = nullptr, *hc3 = nullptr, *hc2 = nullptr, *blob = nullptr;
  const void* leaves[64]; int lsz[64]; int nl = 0;
  for (int i = 0; i < n_in; ++i) {
    int s = in_sizes[i];
    if      (s == Bn * 3 *  256 * WW) hc4 = (const float*)d_in[i];
    else if (s == Bn * 3 * 1024 * WW) hc3 = (const float*)d_in[i];
    else if (s == Bn * 3 * 4096 * WW) hc2 = (const float*)d_in[i];
    else if (s == 357)                blob = (const float*)d_in[i];
    else if (nl < 64) { leaves[nl] = d_in[i]; lsz[nl] = s; ++nl; }
  }
  if (!hc4 || !hc3 || !hc2) return;

  // pipeline order: l4_0,l4_1, l3_0,l3_1, l2_0,l2_1, l43_0,l43_1, l32_0,l32_1
  const int Cin[10] = {3,1, 3,1, 3,1, 2,2, 2,2};
  const int Cou[10] = {1,2, 1,2, 1,2, 2,2, 2,1};
  const int Ksz[10] = {3,3, 3,3, 5,3, 3,3, 3,3};
  // block -> pipeline-layer maps
  const int sortedMap[5][2] = {{4,5},{2,3},{8,9},{0,1},{6,7}}; // l2,l3,l32,l4,l43
  const int insMap[5][2]    = {{0,1},{2,3},{4,5},{6,7},{8,9}}; // l4,l3,l2,l43,l32

  LayerP P[10] = {};
  if (nl >= 40) {
    bool keysSorted   = (lsz[0] <= 2);                       // b,beta,g,w vs w,b,g,beta
    int  wsz0         = keysSorted ? lsz[3] : lsz[0];
    bool blocksSorted = (wsz0 == 75);                        // l2 (k=5) first => sorted
    int li = 0;
    for (int bI = 0; bI < 5; ++bI)
      for (int ly = 0; ly < 2; ++ly) {
        int pi = blocksSorted ? sortedMap[bI][ly] : insMap[bI][ly];
        const float* a0 = (const float*)leaves[li++];
        const float* a1 = (const float*)leaves[li++];
        const float* a2 = (const float*)leaves[li++];
        const float* a3 = (const float*)leaves[li++];
        if (keysSorted) { P[pi].b = a0; P[pi].beta = a1; P[pi].g = a2; P[pi].w = a3; }
        else            { P[pi].w = a0; P[pi].b = a1; P[pi].g = a2; P[pi].beta = a3; }
      }
  } else if (blob) {
    // single flattened blob: assume jax pytree order (sorted keys at every level)
    size_t off = 0;
    for (int bI = 0; bI < 5; ++bI)
      for (int ly = 0; ly < 2; ++ly) {
        int pi = sortedMap[bI][ly];
        int co = Cou[pi];
        int wn = Cou[pi] * Cin[pi] * Ksz[pi] * Ksz[pi];
        P[pi].b    = blob + off; off += co;
        P[pi].beta = blob + off; off += co;
        P[pi].g    = blob + off; off += co;
        P[pi].w    = blob + off; off += wn;
      }
  } else {
    return;
  }

  // ---- workspace layout
  char* ws = (char*)d_ws;
  const size_t SZ_BIG = (size_t)Bn * 2 * 4096 * WW * 4;   // 32 MiB
  float* bufA  = (float*)(ws);
  float* bufB  = (float*)(ws + SZ_BIG);
  float* s4    = (float*)(ws + 2 * SZ_BIG);
  float* s3    = (float*)(ws + 2 * SZ_BIG + (size_t)Bn * 2 *  256 * WW * 4);
  float* s2    = (float*)(ws + 2 * SZ_BIG + (size_t)Bn * 2 * (256 + 1024) * WW * 4);
  float* part  = (float*)(ws + 3 * SZ_BIG + (size_t)Bn * 2 * (256 + 1024) * WW * 4);
  float* stats = part + Bn * NCHUNK * 2;
  (void)ws_size;

  // ---- pipeline
  // l4 branch: [16,3,256,64] -> s4 [16,2,256,64]
  run_layer(hc4, bufA, P[0], Bn, 3, 1, 256, 3, part, stats, stream);
  run_layer(bufA, s4,  P[1], Bn, 1, 2, 256, 3, part, stats, stream);
  // l3 branch: -> s3 [16,2,1024,64]
  run_layer(hc3, bufA, P[2], Bn, 3, 1, 1024, 3, part, stats, stream);
  run_layer(bufA, s3,  P[3], Bn, 1, 2, 1024, 3, part, stats, stream);
  // l2 branch: -> s2 [16,2,4096,64]
  run_layer(hc2, bufA, P[4], Bn, 3, 1, 4096, 5, part, stats, stream);
  run_layer(bufA, s2,  P[5], Bn, 1, 2, 4096, 3, part, stats, stream);

  // merge 4->3: up(s4) + s3 -> bufA ; l43 block -> m43 in bufA
  {
    size_t tot = (size_t)Bn * 2 * 1024 * WW;
    interp_add_kernel<<<(unsigned)((tot + 255) / 256), 256, 0, stream>>>(
        s4, s3, bufA, 256, 1024, 10, tot);
  }
  run_layer(bufA, bufB, P[6], Bn, 2, 2, 1024, 3, part, stats, stream);
  run_layer(bufB, bufA, P[7], Bn, 2, 2, 1024, 3, part, stats, stream);

  // merge 3->2: up(m43) + s2 -> bufB ; l32 block -> d_out [16,1,4096,64]
  {
    size_t tot = (size_t)Bn * 2 * 4096 * WW;
    interp_add_kernel<<<(unsigned)((tot + 255) / 256), 256, 0, stream>>>(
        bufA, s2, bufB, 1024, 4096, 12, tot);
  }
  run_layer(bufB, bufA, P[8], Bn, 2, 2, 4096, 3, part, stats, stream);
  run_layer(bufA, (float*)d_out, P[9], Bn, 2, 1, 4096, 3, part, stats, stream);
  (void)out_size;
}
